// Encoder_79628693668029
// MI455X (gfx1250) — matrix operand order repaired and checked by reading, not compiled, not run on hardware
//
#include <hip/hip_runtime.h>

#define DEV_INLINE __device__ __forceinline__

typedef _Float16 v16h __attribute__((ext_vector_type(16)));
typedef _Float16 v8h  __attribute__((ext_vector_type(8)));
typedef float    v8f  __attribute__((ext_vector_type(8)));

constexpr int N_NODES = 50000;   // multiple of 16 -> 3125 row tiles, no tail
constexpr int N_EDGES = 60000;   // multiple of 16 -> 3750 edge tiles
constexpr int F_IN    = 32;      // == WMMA K exactly
constexpr int DIM     = 64;
constexpr int GDIM    = 3 * DIM; // 192 GRU gate columns
constexpr int NB      = 256;

// ---------------- WMMA helpers (CDNA5 wave32, 16x16x32 f16 -> f32) ----------

DEV_INLINE v8f wmma_fp16(v16h a, v16h b, v8f c) {
  // (neg_a, A, neg_b, B, c_mod, C, reuse_a, reuse_b)
  return __builtin_amdgcn_wmma_f32_16x16x32_f16(false, a, false, b, (short)0, c,
                                                false, false);
}

// A-fragment (16x32 f16, ISA 7.12.2): per lane the 16 elements are two
// contiguous 8-element runs: rowptr[kb .. kb+7] and rowptr[kb+16 .. kb+23],
// kb = (lane>=16 ? 8 : 0) + kofs. Row M = lane & 15.
DEV_INLINE v16h load_a_f16(const _Float16* __restrict__ rowptr, int lane, int kofs) {
  int kb = ((lane >> 4) << 3) + kofs;
  v8h lo = *(const v8h*)(rowptr + kb);
  v8h hi = *(const v8h*)(rowptr + kb + 16);
  v16h a;
#pragma unroll
  for (int e = 0; e < 8; ++e) { a[e] = lo[e]; a[8 + e] = hi[e]; }
  return a;
}

// B-fragment (32x16 f16): column N = lane & 15; elements are K in table order:
// k0+kb+e and k0+kb+16+e (e<8), kb = (lane>=16 ? 8 : 0), from wT[k*ld + n].
DEV_INLINE v16h load_b_f16(const _Float16* __restrict__ wT, int ld, int k0, int n0,
                           int lane) {
  int kb = k0 + ((lane >> 4) << 3);
  const _Float16* p = wT + n0 + (lane & 15);
  v16h b;
#pragma unroll
  for (int e = 0; e < 8; ++e) {
    b[e] = p[(long)(kb + e) * ld];
    b[8 + e] = p[(long)(kb + 16 + e) * ld];
  }
  return b;
}

DEV_INLINE float sigmoidf(float x) { return 1.0f / (1.0f + __expf(-x)); }

DEV_INLINE void atomicMaxF(float* addr, float val) {
  int* ia = reinterpret_cast<int*>(addr);
  int old = __float_as_int(*addr);
  while (__int_as_float(old) < val) {
    int assumed = old;
    old = atomicCAS(ia, assumed, __float_as_int(val));
    if (old == assumed) break;
  }
}

// ---------------- Prep: constant edge matrix W; K-major f16 weight shadows ---

__global__ void prep_weights_kernel(const float* nn1_w, const float* nn1_b,
                                    const float* nn2_w, const float* nn2_b,
                                    const float* lin0_w,
                                    const float* gru_w_ih, const float* gru_w_hh,
                                    _Float16* W16, _Float16* lwT16,
                                    _Float16* ihT16, _Float16* hhT16) {
  __shared__ float hidden[DIM];
  int t = threadIdx.x;
  if (t < DIM) hidden[t] = fmaxf(nn1_w[t] + nn1_b[t], 0.0f);  // edge_attr == 1
  __syncthreads();
  for (int j = t; j < DIM * DIM; j += blockDim.x) {
    float acc = nn2_b[j];
#pragma unroll 8
    for (int d = 0; d < DIM; ++d) acc += hidden[d] * nn2_w[j * DIM + d];
    W16[j] = (_Float16)acc;                       // W[i][o], already K-major
  }
  for (int j = t; j < DIM * F_IN; j += blockDim.x) {
    int n = j / F_IN, k = j % F_IN;               // lin0_w[n][k]
    lwT16[k * DIM + n] = (_Float16)lin0_w[j];
  }
  for (int j = t; j < GDIM * DIM; j += blockDim.x) {
    int n = j / DIM, k = j % DIM;                 // gru_w_*[n][k]
    ihT16[k * GDIM + n] = (_Float16)gru_w_ih[j];
    hhT16[k * GDIM + n] = (_Float16)gru_w_hh[j];
  }
}

__global__ void zero_kernel(float* p, long n) {
  long i = (long)blockIdx.x * blockDim.x + threadIdx.x;
  long stride = (long)gridDim.x * blockDim.x;
  for (; i < n; i += stride) p[i] = 0.0f;
}

__global__ void deg_kernel(const int* edge_index, float* deg) {
  int e = blockIdx.x * blockDim.x + threadIdx.x;
  if (e < N_EDGES) atomicAdd(&deg[edge_index[N_EDGES + e]], 1.0f);
}

__global__ void invdeg_kernel(const float* deg, float* invd) {
  int n = blockIdx.x * blockDim.x + threadIdx.x;
  if (n < N_NODES) {
    float d = deg[n];
    invd[n] = d > 0.0f ? 1.0f / fmaxf(d, 1.0f) : 0.0f;
  }
}

// ---------------- lin0: out = relu(x @ lin0_w.T + b), WMMA K=32 exactly ------

__global__ void lin0_kernel(const float* __restrict__ x,
                            const _Float16* __restrict__ lwT16,
                            const float* __restrict__ lin0_b,
                            float* __restrict__ h, _Float16* __restrict__ h16) {
  int wid = threadIdx.x >> 5, lane = threadIdx.x & 31;
  int tile = blockIdx.x * (blockDim.x >> 5) + wid;
  if (tile >= N_NODES / 16) return;
  int row = tile * 16 + (lane & 15);
  const float* xr = x + (long)row * F_IN;
  int kb = (lane >> 4) << 3;
  v8f xlo = *(const v8f*)(xr + kb);        // 32B-aligned vector loads
  v8f xhi = *(const v8f*)(xr + kb + 16);
  v16h a;
#pragma unroll
  for (int e = 0; e < 8; ++e) { a[e] = (_Float16)xlo[e]; a[8 + e] = (_Float16)xhi[e]; }
  int mbase = (lane >> 4) << 3;
#pragma unroll
  for (int nt = 0; nt < 4; ++nt) {
    v16h b = load_b_f16(lwT16, DIM, 0, nt * 16, lane);
    v8f c = {};
    c = wmma_fp16(a, b, c);
    int col = nt * 16 + (lane & 15);
    float bias = lin0_b[col];
#pragma unroll
    for (int e = 0; e < 8; ++e) {
      long orow = (long)(tile * 16 + mbase + e);
      float v = fmaxf(c[e] + bias, 0.0f);
      h[orow * DIM + col] = v;
      h16[orow * DIM + col] = (_Float16)v;
    }
  }
}

// ---------------- edge conv: gather rows, x W (64x64), scatter-add ----------

__global__ void edge_conv_kernel(const int* __restrict__ edge_index,
                                 const _Float16* __restrict__ h16,
                                 const _Float16* __restrict__ W16,
                                 float* __restrict__ agg) {
  int wid = threadIdx.x >> 5, lane = threadIdx.x & 31;
  int tile = blockIdx.x * (blockDim.x >> 5) + wid;
  if (tile >= N_EDGES / 16) return;
  int src = edge_index[tile * 16 + (lane & 15)];
  const _Float16* hr = h16 + (long)src * DIM;
  v16h a0 = load_a_f16(hr, lane, 0);
  v16h a1 = load_a_f16(hr, lane, 32);
  int mbase = (lane >> 4) << 3;
#pragma unroll
  for (int nt = 0; nt < 4; ++nt) {
    v16h b0 = load_b_f16(W16, DIM, 0, nt * 16, lane);
    v16h b1 = load_b_f16(W16, DIM, 32, nt * 16, lane);
    v8f c = {};
    c = wmma_fp16(a0, b0, c);
    c = wmma_fp16(a1, b1, c);
    int col = nt * 16 + (lane & 15);
#pragma unroll
    for (int e = 0; e < 8; ++e) {
      int dst = edge_index[N_EDGES + tile * 16 + mbase + e];
      atomicAdd(&agg[(long)dst * DIM + col], c[e]);
    }
  }
}

// ---------------- fused GRU: m=relu(agg*invdeg+b); gi,gh GEMMs; gates; h -----

__global__ void gru_kernel(const float* __restrict__ agg,
                           const float* __restrict__ invd,
                           const float* __restrict__ conv_b,
                           const _Float16* __restrict__ ihT16,
                           const _Float16* __restrict__ hhT16,
                           const float* __restrict__ gru_b_ih,
                           const float* __restrict__ gru_b_hh,
                           float* __restrict__ h, _Float16* __restrict__ h16) {
  int wid = threadIdx.x >> 5, lane = threadIdx.x & 31;
  int tile = blockIdx.x * (blockDim.x >> 5) + wid;
  if (tile >= N_NODES / 16) return;
  int row = tile * 16 + (lane & 15);
  float idg = invd[row];
  const float* ar = agg + (long)row * DIM;
  const _Float16* hr = h16 + (long)row * DIM;
  int kb = (lane >> 4) << 3;
  // m = relu(agg*invdeg + conv_b), converted to f16 A-fragments (vector loads)
  v16h am0, am1;
  {
    v8f alo0 = *(const v8f*)(ar + kb);
    v8f ahi0 = *(const v8f*)(ar + kb + 16);
    v8f alo1 = *(const v8f*)(ar + 32 + kb);
    v8f ahi1 = *(const v8f*)(ar + 32 + kb + 16);
    v8f clo0 = *(const v8f*)(conv_b + kb);
    v8f chi0 = *(const v8f*)(conv_b + kb + 16);
    v8f clo1 = *(const v8f*)(conv_b + 32 + kb);
    v8f chi1 = *(const v8f*)(conv_b + 32 + kb + 16);
#pragma unroll
    for (int e = 0; e < 8; ++e) {
      am0[e]     = (_Float16)fmaxf(alo0[e] * idg + clo0[e], 0.0f);
      am0[8 + e] = (_Float16)fmaxf(ahi0[e] * idg + chi0[e], 0.0f);
      am1[e]     = (_Float16)fmaxf(alo1[e] * idg + clo1[e], 0.0f);
      am1[8 + e] = (_Float16)fmaxf(ahi1[e] * idg + chi1[e], 0.0f);
    }
  }
  v16h ah0 = load_a_f16(hr, lane, 0);
  v16h ah1 = load_a_f16(hr, lane, 32);
  int mbase = (lane >> 4) << 3;
#pragma unroll
  for (int t = 0; t < 4; ++t) {
    auto matcol = [&](const _Float16* wT, int n0, v16h x0, v16h x1) -> v8f {
      v16h b0 = load_b_f16(wT, GDIM, 0, n0, lane);
      v16h b1 = load_b_f16(wT, GDIM, 32, n0, lane);
      v8f c = {};
      c = wmma_fp16(x0, b0, c);
      c = wmma_fp16(x1, b1, c);
      return c;
    };
    int col = t * 16 + (lane & 15);
    v8f cir = matcol(ihT16, t * 16, am0, am1);
    v8f chr = matcol(hhT16, t * 16, ah0, ah1);
    v8f ciz = matcol(ihT16, 64 + t * 16, am0, am1);
    v8f chz = matcol(hhT16, 64 + t * 16, ah0, ah1);
    v8f cin = matcol(ihT16, 128 + t * 16, am0, am1);
    v8f chn = matcol(hhT16, 128 + t * 16, ah0, ah1);
    float bir = gru_b_ih[col], bhr = gru_b_hh[col];
    float biz = gru_b_ih[64 + col], bhz = gru_b_hh[64 + col];
    float bin = gru_b_ih[128 + col], bhn = gru_b_hh[128 + col];
#pragma unroll
    for (int e = 0; e < 8; ++e) {
      long orow = (long)(tile * 16 + mbase + e);
      float rg = sigmoidf(cir[e] + bir + chr[e] + bhr);
      float zg = sigmoidf(ciz[e] + biz + chz[e] + bhz);
      float ng = tanhf(cin[e] + bin + rg * (chn[e] + bhn));
      float hold = h[orow * DIM + col];
      float hnew = (1.0f - zg) * ng + zg * hold;
      h[orow * DIM + col] = hnew;
      h16[orow * DIM + col] = (_Float16)hnew;
    }
  }
}

// ---------------- Set2Set ----------------------------------------------------

__global__ void lstm_kernel(const float* __restrict__ w_ih, const float* __restrict__ w_hh,
                            const float* __restrict__ b_ih, const float* __restrict__ b_hh,
                            const float* __restrict__ qstar, float* hl, float* cl,
                            float* rtmp, float* asum, float* emax) {
  int b = blockIdx.x, t = threadIdx.x;
  __shared__ float hsh[DIM];
  __shared__ float qsh[2 * DIM];
  hsh[t] = hl[b * DIM + t];
  qsh[t] = qstar[b * 2 * DIM + t];
  qsh[DIM + t] = qstar[b * 2 * DIM + DIM + t];
  __syncthreads();
  float acc[4];
#pragma unroll
  for (int g = 0; g < 4; ++g) acc[g] = b_ih[g * DIM + t] + b_hh[g * DIM + t];
  for (int k = 0; k < 2 * DIM; ++k) {
    float qv = qsh[k];
#pragma unroll
    for (int g = 0; g < 4; ++g) acc[g] += qv * w_ih[(g * DIM + t) * (2 * DIM) + k];
  }
  for (int k = 0; k < DIM; ++k) {
    float hv = hsh[k];
#pragma unroll
    for (int g = 0; g < 4; ++g) acc[g] += hv * w_hh[(g * DIM + t) * DIM + k];
  }
  float ig = sigmoidf(acc[0]), fg = sigmoidf(acc[1]);
  float gg = tanhf(acc[2]), og = sigmoidf(acc[3]);
  float cnew = fg * cl[b * DIM + t] + ig * gg;
  cl[b * DIM + t] = cnew;
  hl[b * DIM + t] = og * tanhf(cnew);
  // reset this step's attention accumulators
  rtmp[b * DIM + t] = 0.0f;
  if (t == 0) { asum[b] = 0.0f; emax[b] = -1e30f; }
}

__global__ void attn_e_kernel(const float* __restrict__ h, const float* __restrict__ hl,
                              const int* __restrict__ batch, float* e_buf, float* emax) {
  int n = blockIdx.x * blockDim.x + threadIdx.x;
  if (n >= N_NODES) return;
  int b = batch[n];
  const float* hn = h + (long)n * DIM;
  const float* hb = hl + b * DIM;
  float e = 0.0f;
#pragma unroll 8
  for (int k = 0; k < DIM; ++k) e += hn[k] * hb[k];
  e_buf[n] = e;
  atomicMaxF(&emax[b], e);
}

__global__ void attn_acc_kernel(const float* __restrict__ h, const int* __restrict__ batch,
                                const float* __restrict__ e_buf, const float* __restrict__ emax,
                                float* asum, float* rtmp) {
  int n = blockIdx.x, t = threadIdx.x;
  int b = batch[n];
  float em = emax[b];
  if (em < -1e29f) em = 0.0f;                 // mirror isfinite guard
  float p = __expf(e_buf[n] - em);
  if (t == 0) atomicAdd(&asum[b], p);
  atomicAdd(&rtmp[b * DIM + t], p * h[(long)n * DIM + t]);
}

__global__ void s2s_update_kernel(const float* hl, const float* rtmp, const float* asum,
                                  float* qstar) {
  int b = blockIdx.x, t = threadIdx.x;
  float r = rtmp[b * DIM + t] / (asum[b] + 1e-16f);
  qstar[b * 2 * DIM + t] = hl[b * DIM + t];
  qstar[b * 2 * DIM + DIM + t] = r;
}

__global__ void write_out_kernel(const float* qstar, const float* h, float* out) {
  long i = (long)blockIdx.x * blockDim.x + threadIdx.x;
  long stride = (long)gridDim.x * blockDim.x;
  const long q = (long)NB * 2 * DIM;
  const long tot = q + (long)N_NODES * DIM;
  for (; i < tot; i += stride) out[i] = (i < q) ? qstar[i] : h[i - q];
}

// ---------------- host launch ------------------------------------------------

extern "C" void kernel_launch(void* const* d_in, const int* in_sizes, int n_in,
                              void* d_out, int out_size, void* d_ws, size_t ws_size,
                              hipStream_t stream) {
  (void)in_sizes; (void)n_in; (void)out_size; (void)ws_size;
  const float* x         = (const float*)d_in[0];
  const int*   edge_index= (const int*)d_in[1];
  const int*   batch     = (const int*)d_in[2];
  const float* lin0_w    = (const float*)d_in[3];
  const float* lin0_b    = (const float*)d_in[4];
  const float* nn1_w     = (const float*)d_in[5];
  const float* nn1_b     = (const float*)d_in[6];
  const float* nn2_w     = (const float*)d_in[7];
  const float* nn2_b     = (const float*)d_in[8];
  const float* conv_b    = (const float*)d_in[9];
  const float* gru_w_ih  = (const float*)d_in[10];
  const float* gru_w_hh  = (const float*)d_in[11];
  const float* gru_b_ih  = (const float*)d_in[12];
  const float* gru_b_hh  = (const float*)d_in[13];
  const float* lstm_w_ih = (const float*)d_in[14];
  const float* lstm_w_hh = (const float*)d_in[15];
  const float* lstm_b_ih = (const float*)d_in[16];
  const float* lstm_b_hh = (const float*)d_in[17];

  char* ws = (char*)d_ws;
  size_t off = 0;
  auto carve = [&](size_t bytes) -> char* {
    char* p = ws + off;
    off += (bytes + 255) & ~(size_t)255;
    return p;
  };
  _Float16* W16   = (_Float16*)carve((size_t)DIM * DIM * 2);
  _Float16* lwT16 = (_Float16*)carve((size_t)F_IN * DIM * 2);
  _Float16* ihT16 = (_Float16*)carve((size_t)DIM * GDIM * 2);
  _Float16* hhT16 = (_Float16*)carve((size_t)DIM * GDIM * 2);
  float*    h     = (float*)carve((size_t)N_NODES * DIM * 4);
  _Float16* h16   = (_Float16*)carve((size_t)N_NODES * DIM * 2);
  float*    agg   = (float*)carve((size_t)N_NODES * DIM * 4);
  float*    deg   = (float*)carve((size_t)N_NODES * 4);
  float*    invd  = (float*)carve((size_t)N_NODES * 4);
  float*    ebuf  = (float*)carve((size_t)N_NODES * 4);
  float*    qstar = (float*)carve((size_t)NB * 2 * DIM * 4);
  float*    hl    = (float*)carve((size_t)NB * DIM * 4);
  float*    cl    = (float*)carve((size_t)NB * DIM * 4);
  float*    rtmp  = (float*)carve((size_t)NB * DIM * 4);
  float*    asum  = (float*)carve((size_t)NB * 4);
  float*    emax  = (float*)carve((size_t)NB * 4);

  const int rowTiles = N_NODES / 16;   // 3125
  const int edgeTiles = N_EDGES / 16;  // 3750
  const int wavesPerBlock = 8;         // 256 threads = 8 wave32

  prep_weights_kernel<<<1, 256, 0, stream>>>(nn1_w, nn1_b, nn2_w, nn2_b, lin0_w,
                                             gru_w_ih, gru_w_hh, W16, lwT16, ihT16, hhT16);
  zero_kernel<<<256, 256, 0, stream>>>(deg, N_NODES);
  deg_kernel<<<(N_EDGES + 255) / 256, 256, 0, stream>>>(edge_index, deg);
  invdeg_kernel<<<(N_NODES + 255) / 256, 256, 0, stream>>>(deg, invd);

  lin0_kernel<<<(rowTiles + wavesPerBlock - 1) / wavesPerBlock, 256, 0, stream>>>(
      x, lwT16, lin0_b, h, h16);

  for (int it = 0; it < 3; ++it) {
    zero_kernel<<<1024, 256, 0, stream>>>(agg, (long)N_NODES * DIM);
    edge_conv_kernel<<<(edgeTiles + wavesPerBlock - 1) / wavesPerBlock, 256, 0, stream>>>(
        edge_index, h16, W16, agg);
    gru_kernel<<<(rowTiles + wavesPerBlock - 1) / wavesPerBlock, 256, 0, stream>>>(
        agg, invd, conv_b, ihT16, hhT16, gru_b_ih, gru_b_hh, h, h16);
  }

  zero_kernel<<<64, 256, 0, stream>>>(qstar, (long)NB * 2 * DIM);
  zero_kernel<<<64, 256, 0, stream>>>(hl, (long)NB * DIM);
  zero_kernel<<<64, 256, 0, stream>>>(cl, (long)NB * DIM);
  for (int s = 0; s < 3; ++s) {
    lstm_kernel<<<NB, DIM, 0, stream>>>(lstm_w_ih, lstm_w_hh, lstm_b_ih, lstm_b_hh,
                                        qstar, hl, cl, rtmp, asum, emax);
    attn_e_kernel<<<(N_NODES + 255) / 256, 256, 0, stream>>>(h, hl, batch, ebuf, emax);
    attn_acc_kernel<<<N_NODES, DIM, 0, stream>>>(h, batch, ebuf, emax, asum, rtmp);
    s2s_update_kernel<<<NB, DIM, 0, stream>>>(hl, rtmp, asum, qstar);
  }

  write_out_kernel<<<2048, 256, 0, stream>>>(qstar, h, (float*)d_out);
}
